// GNNDecoder_68659347194441
// MI455X (gfx1250) — matrix operand order repaired and checked by reading, not compiled, not run on hardware
//
#include <hip/hip_runtime.h>
#include <hip/hip_bf16.h>

// ---------------------------------------------------------------------------
// GNN decoder for MI455X (gfx1250), wave32 + WMMA bf16 (f32 accumulate).
//
//   per layer:  h   = relu(x @ W_lin + b_lin)                 [gemm_relu]
//               msg = relu([h[src], ea] @ W_msg + b_msg)      [edge_msg]
//               agg = segment_sum(msg, dst)                   [fused atomics]
//               x   = relu(agg*inv_deg @ W_l + b_l + x @ W_r) [node_update]
//   emb    = relu(x @ W_dec + b_dec)                          [gemm_relu]
//   logits = emb @ W_fin + b_fin                              [gemm_final]
//
// v5: add per-iteration compiler memory barrier in the persistent loops so
// LICM cannot hoist the LDS B-fragment loads (which caused 256-VGPR register
// pressure and scratch spill/reload of every B fragment in v4).
// ---------------------------------------------------------------------------

typedef __attribute__((ext_vector_type(16))) __bf16 v16bf;
typedef __attribute__((ext_vector_type(8)))  float  v8f;

#define HDIM 128
#define WPB  8          // waves per block (block = 256 threads, wave32)
#define TB   256

#if defined(__has_builtin)
#if __has_builtin(__builtin_amdgcn_global_load_async_to_lds_b128) && \
    __has_builtin(__builtin_amdgcn_s_wait_asynccnt)
#define USE_ASYNC_LDS 1
#endif
#endif

// prevents loop-invariant code motion of LDS reads across iterations
#define LOOP_FENCE() asm volatile("" ::: "memory")

__device__ __forceinline__ unsigned short f2bf(float f) {
    unsigned int u = __float_as_uint(f);
    unsigned int r = (u + 0x7FFFu + ((u >> 16) & 1u)) >> 16;   // RNE bf16
    return (unsigned short)r;
}

union FragA { v16bf v; unsigned short u[16]; };

// A fragment (16x32 bf16) from row-major bf16 row pointer, ld = HDIM.
__device__ __forceinline__ v16bf load_a_frag(const unsigned short* rowp, int kt, int hi) {
    v16bf a;
    const unsigned short* p = rowp + kt * 32 + hi * 8;
    *((uint4*)&a)     = *(const uint4*)(p);
    *((uint4*)&a + 1) = *(const uint4*)(p + 16);
    return a;
}

// B fragment (32x16 bf16) from transposed weight Wt[N][K] (K contiguous).
__device__ __forceinline__ v16bf load_b_frag(const unsigned short* wt, int nt, int kt,
                                             int l15, int hi) {
    const unsigned short* p = wt + (size_t)(nt * 16 + l15) * HDIM + kt * 32 + hi * 8;
    v16bf b;
    *((uint4*)&b)     = *(const uint4*)(p);
    *((uint4*)&b + 1) = *(const uint4*)(p + 16);
    return b;
}

// cooperative global -> LDS stage (256 threads, 16B vectors)
__device__ __forceinline__ void stage_lds(unsigned short* dst, const unsigned short* src,
                                          int nelem) {
    int nv = nelem >> 3;
#ifdef USE_ASYNC_LDS
    typedef int v4i_vs __attribute__((vector_size(16)));
    typedef __attribute__((address_space(1))) v4i_vs* gptr_t;
    typedef __attribute__((address_space(3))) v4i_vs* lptr_t;
    gptr_t g = (gptr_t)(unsigned short*)src;
    lptr_t l = (lptr_t)dst;
    for (int i = threadIdx.x; i < nv; i += TB)
        __builtin_amdgcn_global_load_async_to_lds_b128(g + i, l + i, 0, 0);
    __builtin_amdgcn_s_wait_asynccnt(0);
#else
    const uint4* s = (const uint4*)src;
    uint4* d = (uint4*)dst;
    for (int i = threadIdx.x; i < nv; i += TB) d[i] = s[i];
#endif
}

// ---------------------------------------------------------------------------
// small prep kernels
// ---------------------------------------------------------------------------
__global__ void k_zero_f32(float* p, int n) {
    int i = blockIdx.x * blockDim.x + threadIdx.x;
    if (i < n) p[i] = 0.0f;
}

__global__ void k_f32_to_bf16(const float* __restrict__ src, unsigned short* __restrict__ dst, int n) {
    int i = blockIdx.x * blockDim.x + threadIdx.x;
    if (i < n) dst[i] = f2bf(src[i]);
}

// Wt[n*K + k] = bf16(W[k*Nn + n])
__global__ void k_transpose_bf16(const float* __restrict__ W, unsigned short* __restrict__ Wt,
                                 int K, int Nn) {
    int i = blockIdx.x * blockDim.x + threadIdx.x;
    if (i >= K * Nn) return;
    int n = i / K, k = i - n * K;
    Wt[i] = f2bf(W[(size_t)k * Nn + n]);
}

__global__ void k_degree(const int* __restrict__ dst, float* __restrict__ deg, int E) {
    int i = blockIdx.x * blockDim.x + threadIdx.x;
    if (i < E) atomicAdd(&deg[dst[i]], 1.0f);
}

__global__ void k_invdeg(float* __restrict__ deg, int N) {
    int i = blockIdx.x * blockDim.x + threadIdx.x;
    if (i < N) deg[i] = 1.0f / fmaxf(deg[i], 1.0f);
}

// ---------------------------------------------------------------------------
// Y = relu(X @ Wt^T + bias). Persistent: one LDS stage per block, grid-stride
// over 16-row M-tiles (one wave per tile iteration).
// ---------------------------------------------------------------------------
__global__ void k_gemm_relu(const unsigned short* __restrict__ X,
                            const unsigned short* __restrict__ Wt,
                            const float* __restrict__ bias,
                            unsigned short* __restrict__ Y,
                            int M, int Mtiles) {
    __shared__ __align__(32) unsigned short lds_w[HDIM * HDIM];
    stage_lds(lds_w, Wt, HDIM * HDIM);

    int lane = threadIdx.x & 31;
    int l15 = lane & 15, hi = lane >> 4;
    int wid0   = (blockIdx.x * blockDim.x + threadIdx.x) >> 5;
    int wtotal = (gridDim.x * blockDim.x) >> 5;

    float bcol[8];
#pragma unroll
    for (int nt = 0; nt < 8; ++nt) bcol[nt] = bias[nt * 16 + l15];

    __syncthreads();

    for (int t = wid0; t < Mtiles; t += wtotal) {
        LOOP_FENCE();
        int m0 = t * 16;
        int mrow = m0 + l15; if (mrow >= M) mrow = M - 1;
        const unsigned short* arow = X + (size_t)mrow * HDIM;

        v16bf afr[4];
#pragma unroll
        for (int kt = 0; kt < 4; ++kt) afr[kt] = load_a_frag(arow, kt, hi);

        v8f acc[8];
#pragma unroll
        for (int nt = 0; nt < 8; ++nt)
#pragma unroll
            for (int v = 0; v < 8; ++v) acc[nt][v] = 0.0f;

#pragma unroll
        for (int kt = 0; kt < 4; ++kt) {
            v16bf bfr[8];
#pragma unroll
            for (int nt = 0; nt < 8; ++nt) bfr[nt] = load_b_frag(lds_w, nt, kt, l15, hi);
#pragma unroll
            for (int nt = 0; nt < 8; ++nt)
                acc[nt] = __builtin_amdgcn_wmma_f32_16x16x32_bf16(
                    false, afr[kt], false, bfr[nt], (short)0, acc[nt], false, false);
        }

#pragma unroll
        for (int nt = 0; nt < 8; ++nt)
#pragma unroll
            for (int v = 0; v < 8; ++v) {
                int r = m0 + v + 8 * hi;
                if (r < M)
                    Y[(size_t)r * HDIM + nt * 16 + l15] =
                        f2bf(fmaxf(acc[nt][v] + bcol[nt], 0.0f));
            }
    }
}

// ---------------------------------------------------------------------------
// logits = emb @ Wfin^T + b_fin  (f32 out, OUT=64 -> 4 N-tiles, no relu)
// ---------------------------------------------------------------------------
__global__ void k_gemm_final(const unsigned short* __restrict__ X,
                             const unsigned short* __restrict__ Wt,
                             const float* __restrict__ bias,
                             float* __restrict__ Y,
                             int M, int Mtiles) {
    __shared__ __align__(32) unsigned short lds_w[64 * HDIM];
    stage_lds(lds_w, Wt, 64 * HDIM);

    int lane = threadIdx.x & 31;
    int l15 = lane & 15, hi = lane >> 4;
    int wid0   = (blockIdx.x * blockDim.x + threadIdx.x) >> 5;
    int wtotal = (gridDim.x * blockDim.x) >> 5;

    float bcol[4];
#pragma unroll
    for (int nt = 0; nt < 4; ++nt) bcol[nt] = bias[nt * 16 + l15];

    __syncthreads();

    for (int t = wid0; t < Mtiles; t += wtotal) {
        LOOP_FENCE();
        int m0 = t * 16;
        int mrow = m0 + l15; if (mrow >= M) mrow = M - 1;
        const unsigned short* arow = X + (size_t)mrow * HDIM;

        v16bf afr[4];
#pragma unroll
        for (int kt = 0; kt < 4; ++kt) afr[kt] = load_a_frag(arow, kt, hi);

        v8f acc[4];
#pragma unroll
        for (int nt = 0; nt < 4; ++nt)
#pragma unroll
            for (int v = 0; v < 8; ++v) acc[nt][v] = 0.0f;

#pragma unroll
        for (int kt = 0; kt < 4; ++kt) {
            v16bf bfr[4];
#pragma unroll
            for (int nt = 0; nt < 4; ++nt) bfr[nt] = load_b_frag(lds_w, nt, kt, l15, hi);
#pragma unroll
            for (int nt = 0; nt < 4; ++nt)
                acc[nt] = __builtin_amdgcn_wmma_f32_16x16x32_bf16(
                    false, afr[kt], false, bfr[nt], (short)0, acc[nt], false, false);
        }

#pragma unroll
        for (int nt = 0; nt < 4; ++nt)
#pragma unroll
            for (int v = 0; v < 8; ++v) {
                int r = m0 + v + 8 * hi;
                if (r < M) Y[(size_t)r * 64 + nt * 16 + l15] = acc[nt][v] + bcol[nt];
            }
    }
}

// ---------------------------------------------------------------------------
// Fused edge kernel (persistent): per 16-edge tile,
//   msg = relu(h[src] @ Wmsg_t + ea*Wmsg_row128 + b_msg); agg[dst] += msg
// ---------------------------------------------------------------------------
__global__ void k_edge_msg(const unsigned short* __restrict__ Hb,
                           const float* __restrict__ ea,
                           const int* __restrict__ src,
                           const int* __restrict__ dst,
                           const unsigned short* __restrict__ Wt,
                           const float* __restrict__ wrow128,
                           const float* __restrict__ bias,
                           float* __restrict__ agg,
                           int E, int Etiles) {
    __shared__ __align__(32) unsigned short lds_w[HDIM * HDIM];
    stage_lds(lds_w, Wt, HDIM * HDIM);

    int lane = threadIdx.x & 31;
    int l15 = lane & 15, hi = lane >> 4;
    int wid0   = (blockIdx.x * blockDim.x + threadIdx.x) >> 5;
    int wtotal = (gridDim.x * blockDim.x) >> 5;

    float wv[8], bv[8];
#pragma unroll
    for (int nt = 0; nt < 8; ++nt) {
        int c = nt * 16 + l15;
        wv[nt] = wrow128[c];
        bv[nt] = bias[c];
    }

    __syncthreads();

    for (int t = wid0; t < Etiles; t += wtotal) {
        LOOP_FENCE();
        int e0 = t * 16;
        int er = e0 + l15; if (er >= E) er = E - 1;
        const unsigned short* arow = Hb + (size_t)src[er] * HDIM;

        v16bf afr[4];
#pragma unroll
        for (int kt = 0; kt < 4; ++kt) afr[kt] = load_a_frag(arow, kt, hi);

        float ea8[8];
        int   drow[8];
#pragma unroll
        for (int v = 0; v < 8; ++v) {
            int r = e0 + v + 8 * hi;
            int rc = r < E ? r : E - 1;
            ea8[v]  = ea[rc];
            drow[v] = dst[rc];
        }

        v8f acc[8];
#pragma unroll
        for (int nt = 0; nt < 8; ++nt)
#pragma unroll
            for (int v = 0; v < 8; ++v) acc[nt][v] = 0.0f;

#pragma unroll
        for (int kt = 0; kt < 4; ++kt) {
            v16bf bfr[8];
#pragma unroll
            for (int nt = 0; nt < 8; ++nt) bfr[nt] = load_b_frag(lds_w, nt, kt, l15, hi);
#pragma unroll
            for (int nt = 0; nt < 8; ++nt)
                acc[nt] = __builtin_amdgcn_wmma_f32_16x16x32_bf16(
                    false, afr[kt], false, bfr[nt], (short)0, acc[nt], false, false);
        }

#pragma unroll
        for (int nt = 0; nt < 8; ++nt) {
            int c = nt * 16 + l15;
#pragma unroll
            for (int v = 0; v < 8; ++v) {
                int r = e0 + v + 8 * hi;
                if (r < E) {
                    float val = fmaxf(acc[nt][v] + fmaf(ea8[v], wv[nt], bv[nt]), 0.0f);
                    atomicAdd(&agg[(size_t)drow[v] * HDIM + c], val);
                }
            }
        }
    }
}

// ---------------------------------------------------------------------------
// Node update (persistent): xnew = relu(agg*inv_deg @ Wl_t + b_l + x @ Wr_t)
// ---------------------------------------------------------------------------
__global__ void k_node_update(const float* __restrict__ agg,
                              const float* __restrict__ inv,
                              const unsigned short* __restrict__ X,
                              const unsigned short* __restrict__ Wl_t,
                              const unsigned short* __restrict__ Wr_t,
                              const float* __restrict__ bias,
                              unsigned short* __restrict__ Y,
                              int M, int Mtiles) {
    __shared__ __align__(32) unsigned short lds_wl[HDIM * HDIM];
    __shared__ __align__(32) unsigned short lds_wr[HDIM * HDIM];
    stage_lds(lds_wl, Wl_t, HDIM * HDIM);
    stage_lds(lds_wr, Wr_t, HDIM * HDIM);

    int lane = threadIdx.x & 31;
    int l15 = lane & 15, hi = lane >> 4;
    int wid0   = (blockIdx.x * blockDim.x + threadIdx.x) >> 5;
    int wtotal = (gridDim.x * blockDim.x) >> 5;

    float bcol[8];
#pragma unroll
    for (int nt = 0; nt < 8; ++nt) bcol[nt] = bias[nt * 16 + l15];

    __syncthreads();

    for (int t = wid0; t < Mtiles; t += wtotal) {
        LOOP_FENCE();
        int m0 = t * 16;
        int mrow = m0 + l15; if (mrow >= M) mrow = M - 1;
        float s = inv[mrow];
        const float* aggrow = agg + (size_t)mrow * HDIM;
        const unsigned short* xrow = X + (size_t)mrow * HDIM;

        // scaled bf16 A fragments from f32 agg + hoisted x fragments
        FragA af[4];
#pragma unroll
        for (int kt = 0; kt < 4; ++kt) {
            const float* p = aggrow + kt * 32 + hi * 8;
#pragma unroll
            for (int i = 0; i < 8; ++i) af[kt].u[i]     = f2bf(p[i] * s);
#pragma unroll
            for (int i = 0; i < 8; ++i) af[kt].u[8 + i] = f2bf(p[16 + i] * s);
        }
        v16bf xf[4];
#pragma unroll
        for (int kt = 0; kt < 4; ++kt) xf[kt] = load_a_frag(xrow, kt, hi);

        v8f acc[8];
#pragma unroll
        for (int nt = 0; nt < 8; ++nt)
#pragma unroll
            for (int v = 0; v < 8; ++v) acc[nt][v] = 0.0f;

#pragma unroll
        for (int kt = 0; kt < 4; ++kt) {
            v16bf bfr[8];
#pragma unroll
            for (int nt = 0; nt < 8; ++nt) bfr[nt] = load_b_frag(lds_wl, nt, kt, l15, hi);
#pragma unroll
            for (int nt = 0; nt < 8; ++nt)
                acc[nt] = __builtin_amdgcn_wmma_f32_16x16x32_bf16(
                    false, af[kt].v, false, bfr[nt], (short)0, acc[nt], false, false);
        }
#pragma unroll
        for (int kt = 0; kt < 4; ++kt) {
            v16bf bfr[8];
#pragma unroll
            for (int nt = 0; nt < 8; ++nt) bfr[nt] = load_b_frag(lds_wr, nt, kt, l15, hi);
#pragma unroll
            for (int nt = 0; nt < 8; ++nt)
                acc[nt] = __builtin_amdgcn_wmma_f32_16x16x32_bf16(
                    false, xf[kt], false, bfr[nt], (short)0, acc[nt], false, false);
        }

#pragma unroll
        for (int nt = 0; nt < 8; ++nt)
#pragma unroll
            for (int v = 0; v < 8; ++v) {
                int r = m0 + v + 8 * hi;
                if (r < M)
                    Y[(size_t)r * HDIM + nt * 16 + l15] =
                        f2bf(fmaxf(acc[nt][v] + bcol[nt], 0.0f));
            }
    }
}

// ---------------------------------------------------------------------------
// host-side orchestration
// ---------------------------------------------------------------------------
static inline int imin(int a, int b) { return a < b ? a : b; }

extern "C" void kernel_launch(void* const* d_in, const int* in_sizes, int n_in,
                              void* d_out, int out_size, void* d_ws, size_t ws_size,
                              hipStream_t stream) {
    const float* x_f   = (const float*)d_in[0];
    const float* eattr = (const float*)d_in[1];
    const float* W_lin = (const float*)d_in[2];
    const float* b_lin = (const float*)d_in[3];
    const float* W_msg = (const float*)d_in[4];
    const float* b_msg = (const float*)d_in[5];
    const float* W_l   = (const float*)d_in[6];
    const float* b_l   = (const float*)d_in[7];
    const float* W_r   = (const float*)d_in[8];
    const float* W_dec = (const float*)d_in[9];
    const float* b_dec = (const float*)d_in[10];
    const float* W_fin = (const float*)d_in[11];
    const float* b_fin = (const float*)d_in[12];
    const int*   eidx  = (const int*)d_in[13];

    const int Hd = HDIM, Ly = 3;
    const int N = in_sizes[0] / Hd;
    const int E = in_sizes[1];
    const int* srcI = eidx;
    const int* dstI = eidx + E;

    // workspace carve (256B aligned)
    char* base = (char*)d_ws;
    size_t off = 0;
    auto carve = [&](size_t bytes) -> void* {
        void* r = base + off;
        off = (off + bytes + 255) & ~(size_t)255;
        return r;
    };
    unsigned short* xa     = (unsigned short*)carve((size_t)N * Hd * 2);
    unsigned short* xb     = (unsigned short*)carve((size_t)N * Hd * 2);
    unsigned short* hbuf   = (unsigned short*)carve((size_t)N * Hd * 2);   // h / emb
    float*          agg    = (float*)carve((size_t)N * Hd * 4);
    float*          invd   = (float*)carve((size_t)N * 4);
    unsigned short* Wlin_t = (unsigned short*)carve((size_t)Ly * Hd * Hd * 2);
    unsigned short* Wmsg_t = (unsigned short*)carve((size_t)Ly * Hd * Hd * 2);
    unsigned short* Wl_t   = (unsigned short*)carve((size_t)Ly * Hd * Hd * 2);
    unsigned short* Wr_t   = (unsigned short*)carve((size_t)Ly * Hd * Hd * 2);
    unsigned short* Wdec_t = (unsigned short*)carve((size_t)Hd * Hd * 2);
    unsigned short* Wfin_t = (unsigned short*)carve((size_t)64 * Hd * 2);

    const int wblk = (Hd * Hd + TB - 1) / TB;

    // ---- weight transpose + bf16 conversion ----
    for (int l = 0; l < Ly; ++l) {
        k_transpose_bf16<<<wblk, TB, 0, stream>>>(W_lin + (size_t)l * Hd * Hd, Wlin_t + (size_t)l * Hd * Hd, Hd, Hd);
        k_transpose_bf16<<<wblk, TB, 0, stream>>>(W_msg + (size_t)l * (Hd + 1) * Hd, Wmsg_t + (size_t)l * Hd * Hd, Hd, Hd);
        k_transpose_bf16<<<wblk, TB, 0, stream>>>(W_l + (size_t)l * Hd * Hd, Wl_t + (size_t)l * Hd * Hd, Hd, Hd);
        k_transpose_bf16<<<wblk, TB, 0, stream>>>(W_r + (size_t)l * Hd * Hd, Wr_t + (size_t)l * Hd * Hd, Hd, Hd);
    }
    k_transpose_bf16<<<wblk, TB, 0, stream>>>(W_dec, Wdec_t, Hd, Hd);
    k_transpose_bf16<<<(Hd * 64 + TB - 1) / TB, TB, 0, stream>>>(W_fin, Wfin_t, Hd, 64);

    // ---- x -> bf16 ----
    k_f32_to_bf16<<<((size_t)N * Hd + TB - 1) / TB, TB, 0, stream>>>(x_f, xa, N * Hd);

    // ---- degree -> inverse degree ----
    k_zero_f32<<<(N + TB - 1) / TB, TB, 0, stream>>>(invd, N);
    k_degree<<<(E + TB - 1) / TB, TB, 0, stream>>>(dstI, invd, E);
    k_invdeg<<<(N + TB - 1) / TB, TB, 0, stream>>>(invd, N);

    const int Mtiles = (N + 15) / 16;
    const int Etiles = (E + 15) / 16;
    const int gemmBlocks = imin((Mtiles + WPB - 1) / WPB, 512);
    const int edgeBlocks = imin((Etiles + WPB - 1) / WPB, 1024);

    unsigned short* xcur = xa;
    unsigned short* xnxt = xb;
    for (int l = 0; l < Ly; ++l) {
        // h = relu(x @ W_lin + b_lin)
        k_gemm_relu<<<gemmBlocks, TB, 0, stream>>>(
            xcur, Wlin_t + (size_t)l * Hd * Hd, b_lin + (size_t)l * Hd, hbuf, N, Mtiles);
        // agg = 0
        k_zero_f32<<<((size_t)N * Hd + TB - 1) / TB, TB, 0, stream>>>(agg, N * Hd);
        // edge messages + scatter-add
        k_edge_msg<<<edgeBlocks, TB, 0, stream>>>(
            hbuf, eattr, srcI, dstI,
            Wmsg_t + (size_t)l * Hd * Hd,
            W_msg + (size_t)l * (Hd + 1) * Hd + (size_t)Hd * Hd,
            b_msg + (size_t)l * Hd, agg, E, Etiles);
        // x = relu(agg*inv @ W_l + b_l + x @ W_r)
        k_node_update<<<gemmBlocks, TB, 0, stream>>>(
            agg, invd, xcur,
            Wl_t + (size_t)l * Hd * Hd, Wr_t + (size_t)l * Hd * Hd,
            b_l + (size_t)l * Hd, xnxt, N, Mtiles);
        unsigned short* t = xcur; xcur = xnxt; xnxt = t;
    }

    // emb = relu(x @ W_dec + b_dec)
    k_gemm_relu<<<gemmBlocks, TB, 0, stream>>>(xcur, Wdec_t, b_dec, hbuf, N, Mtiles);
    // logits = emb @ W_fin + b_fin
    k_gemm_final<<<gemmBlocks, TB, 0, stream>>>(hbuf, Wfin_t, b_fin, (float*)d_out, N, Mtiles);
}
